// Net_27307402068116
// MI455X (gfx1250) — compile-verified, matrix operand order checked by reading, not run on hardware
//
#include <hip/hip_runtime.h>
#include <hip/hip_bf16.h>

typedef __attribute__((ext_vector_type(16))) _Float16 v16h;
typedef __attribute__((ext_vector_type(8)))  _Float16 h8;
typedef __attribute__((ext_vector_type(8)))  float    v8f;

#define NN      256
#define DIMM    128
#define NF      127
#define HH      8
#define DHH     64
#define INNER   512
#define DEPTH   6
#define EDIM    3
#define NB      512
#define SCALE   0.125f
#define EPS     1e-5f

#define WMMA16(a,b,c) __builtin_amdgcn_wmma_f32_16x16x32_f16(false,(a),false,(b),(short)0,(c),false,false)

// ---- CDNA5 async memory->LDS (ASYNCcnt) -----------------------------------
__device__ inline void async_copy_b128(void* lds_dst, const void* gsrc) {
  uint32_t off = (uint32_t)(uintptr_t)lds_dst;   // low 32 bits = LDS byte offset
  asm volatile("global_load_async_to_lds_b128 %0, %1, off"
               :: "v"(off), "v"(gsrc) : "memory");
}
__device__ inline void async_wait0() {
  asm volatile("s_wait_asynccnt 0x0" ::: "memory");
}

// ---- wave32 helpers -------------------------------------------------------
__device__ inline float wred_sum(float v) {
  #pragma unroll
  for (int o = 16; o > 0; o >>= 1) v += __shfl_xor(v, o, 32);
  return v;
}
__device__ inline float wred_max(float v) {
  #pragma unroll
  for (int o = 16; o > 0; o >>= 1) v = fmaxf(v, __shfl_xor(v, o, 32));
  return v;
}

// Fragment layout (ISA 7.12.2, f16 16x16x32): for a lane with half-select `hi`,
// the 16 halves are two contiguous 8-half runs: [k0+hi*8, +8) and [k0+16+hi*8, +8).
__device__ inline v16h frag_2x128(const _Float16* p) {
  union { v16h v; h8 h[2]; } u;
  u.h[0] = *(const h8*)(p);
  u.h[1] = *(const h8*)(p + 16);
  return u.v;
}
// A fragment (16x32 MxK) from row-major f16 (LDS), leading dim ld (mult of 8)
__device__ inline v16h frag_a(const _Float16* s, int ld, int k0) {
  int lane = threadIdx.x & 31, m = lane & 15, hi = lane >> 4;
  return frag_2x128(s + m * ld + k0 + hi * 8);
}
// B fragment (32x16 KxN) from *transposed* f16 storage Wt[n][k], leading dim ld
__device__ inline v16h frag_b_t(const _Float16* Wt, int ld, int k0, int n0) {
  int lane = threadIdx.x & 31, nn = lane & 15, hi = lane >> 4;
  return frag_2x128(Wt + (n0 + nn) * ld + k0 + hi * 8);
}
// Store 16x16 f32 C fragment to LDS row-major, leading dim ld
__device__ inline void store_c_lds(float* s, int ld, v8f c) {
  int lane = threadIdx.x & 31, nn = lane & 15, hi = lane >> 4;
  union { v8f v; float f[8]; } u; u.v = c;
  #pragma unroll
  for (int r = 0; r < 8; ++r) s[(r + hi * 8) * ld + nn] = u.f[r];
}

// ---- kernels --------------------------------------------------------------

__global__ void k_build_nodes(const float* __restrict__ atom_emb,
                              const float* __restrict__ noise,
                              float* __restrict__ nodes) {
  int idx = blockIdx.x * blockDim.x + threadIdx.x;
  if (idx >= NN * DIMM) return;
  int i = idx >> 7, c = idx & 127;
  nodes[idx] = (c < NF) ? atom_emb[i * NF + c] : noise[0];
}

// Per-layer weight prep: f32 -> f16 with transpose (Wt[n*K + k] = W[k*N + n]).
// Segments: WqT(K128,N512) WkvT(K128,N1024) WoT(K512,N128) W1T(K128,N512) W2T(K512,N128)
__global__ void k_prep(const float* __restrict__ Wq, const float* __restrict__ Wkv,
                       const float* __restrict__ Wo, const float* __restrict__ W1,
                       const float* __restrict__ W2,
                       _Float16* __restrict__ WqT, _Float16* __restrict__ WkvT,
                       _Float16* __restrict__ WoT, _Float16* __restrict__ W1T,
                       _Float16* __restrict__ W2T) {
  int idx = blockIdx.x * 256 + threadIdx.x;
  const float* W; _Float16* T; int s, Nc, off;
  if      (idx <  65536) { W = Wq;  T = WqT;  s = 7; Nc = 512;  off = 0; }
  else if (idx < 196608) { W = Wkv; T = WkvT; s = 7; Nc = 1024; off = 65536; }
  else if (idx < 262144) { W = Wo;  T = WoT;  s = 9; Nc = 128;  off = 196608; }
  else if (idx < 327680) { W = W1;  T = W1T;  s = 7; Nc = 512;  off = 262144; }
  else if (idx < 393216) { W = W2;  T = W2T;  s = 9; Nc = 128;  off = 327680; }
  else return;
  int li = idx - off, n = li >> s, k = li & ((1 << s) - 1);
  T[li] = (_Float16)W[k * Nc + n];
}

// LayerNorm1 + Q/K/V projections. q16 gets SCALE folded; k'/v' get bkv+be folded.
// Also emits d-major transposed copy of v' for the P@V' WMMA.
__global__ void k_ln_qkv(const float* __restrict__ nodes,
                         const float* __restrict__ lg, const float* __restrict__ lb,
                         const _Float16* __restrict__ WqT, const float* __restrict__ bq,
                         const _Float16* __restrict__ WkvT, const float* __restrict__ bkv,
                         const float* __restrict__ be,
                         _Float16* __restrict__ q16, _Float16* __restrict__ k16,
                         _Float16* __restrict__ v16T) {
  __shared__ __align__(16) float    Xs[16 * DIMM];
  __shared__ __align__(16) _Float16 Xh[16 * DIMM];
  int i0 = blockIdx.x * 16, t = threadIdx.x, wave = t >> 5, lane = t & 31;
  for (int idx = t; idx < 16 * DIMM / 4; idx += 256) {         // async f32 tile -> LDS
    int r = idx >> 5, c4 = (idx & 31) * 4;
    async_copy_b128(&Xs[r * DIMM + c4], &nodes[(i0 + r) * DIMM + c4]);
  }
  async_wait0();
  __syncthreads();
  #pragma unroll
  for (int rr = 0; rr < 2; ++rr) {
    int r = wave * 2 + rr;
    float s = 0.f;
    for (int c = lane; c < DIMM; c += 32) s += Xs[r * DIMM + c];
    float mean = wred_sum(s) * (1.f / DIMM);
    float vs = 0.f;
    for (int c = lane; c < DIMM; c += 32) { float d = Xs[r * DIMM + c] - mean; vs += d * d; }
    float inv = rsqrtf(wred_sum(vs) * (1.f / DIMM) + EPS);
    for (int c = lane; c < DIMM; c += 32)
      Xh[r * DIMM + c] = (_Float16)((Xs[r * DIMM + c] - mean) * inv * lg[c] + lb[c]);
  }
  __syncthreads();

  int nn = lane & 15, hi = lane >> 4;
  // ---- q tiles (cols 0..511 of Wq), uniform store path
  for (int tt = wave; tt < 32; tt += 8) {
    int n0 = tt * 16;
    __builtin_prefetch(&WqT[n0 * DIMM], 0, 1);
    v8f c = {};
    #pragma unroll
    for (int kc = 0; kc < 4; ++kc)
      c = WMMA16(frag_a(Xh, DIMM, kc * 32), frag_b_t(WqT, DIMM, kc * 32, n0), c);
    union { v8f v; float f[8]; } u; u.v = c;
    #pragma unroll
    for (int r8 = 0; r8 < 8; ++r8) {
      int row = i0 + r8 + hi * 8, col = n0 + nn;
      q16[row * INNER + col] = (_Float16)((u.f[r8] + bq[col]) * SCALE);
    }
  }
  // ---- k tiles (cols 0..511 of Wkv)
  for (int tt = wave; tt < 32; tt += 8) {
    int n0 = tt * 16;
    __builtin_prefetch(&WkvT[n0 * DIMM], 0, 1);
    v8f c = {};
    #pragma unroll
    for (int kc = 0; kc < 4; ++kc)
      c = WMMA16(frag_a(Xh, DIMM, kc * 32), frag_b_t(WkvT, DIMM, kc * 32, n0), c);
    union { v8f v; float f[8]; } u; u.v = c;
    #pragma unroll
    for (int r8 = 0; r8 < 8; ++r8) {
      int row = i0 + r8 + hi * 8, col = n0 + nn;
      k16[row * INNER + col] = (_Float16)(u.f[r8] + bkv[col] + be[col]);
    }
  }
  // ---- v tiles (cols 512..1023 of Wkv) -> d-major transposed store
  for (int tt = wave; tt < 32; tt += 8) {
    int n0 = INNER + tt * 16;
    __builtin_prefetch(&WkvT[n0 * DIMM], 0, 1);
    v8f c = {};
    #pragma unroll
    for (int kc = 0; kc < 4; ++kc)
      c = WMMA16(frag_a(Xh, DIMM, kc * 32), frag_b_t(WkvT, DIMM, kc * 32, n0), c);
    union { v8f v; float f[8]; } u; u.v = c;
    #pragma unroll
    for (int r8 = 0; r8 < 8; ++r8) {
      int row = i0 + r8 + hi * 8, col = n0 + nn, cv = col - INNER;
      v16T[cv * NN + row] = (_Float16)(u.f[r8] + bkv[col] + be[cv]);
    }
  }
}

// Attention for one (head, 16-row query tile): dense WMMA + sparse bond terms.
__global__ void k_attn(const _Float16* __restrict__ q16, const _Float16* __restrict__ k16,
                       const _Float16* __restrict__ v16T, const float* __restrict__ We,
                       const int* __restrict__ bonds, const float* __restrict__ coords,
                       _Float16* __restrict__ attn16) {
  __shared__ __align__(16) _Float16 Qs[16 * DHH];
  __shared__ __align__(16) float    Ss[16 * NN];
  __shared__ __align__(16) _Float16 P16[16 * NN];
  __shared__ __align__(16) float    Os[16 * DHH];
  __shared__ float    WeH[EDIM * DHH];
  int h = blockIdx.x >> 4, it = blockIdx.x & 15, i0 = it * 16;
  int t = threadIdx.x, wave = t >> 5, lane = t & 31;

  for (int idx = t; idx < 16 * DHH / 8; idx += 256) {          // async Q tile -> LDS
    int r = idx >> 3, c8 = (idx & 7) * 8;
    async_copy_b128(&Qs[r * DHH + c8], &q16[(i0 + r) * INNER + h * DHH + c8]);
  }
  for (int idx = t; idx < EDIM * DHH; idx += 256)
    WeH[idx] = We[(idx / DHH) * INNER + h * DHH + (idx & 63)];
  async_wait0();
  __syncthreads();

  // S = (q*SCALE) @ K'^T   (16 x 256); K' row-major [j][d] is already B-friendly
  for (int jt = wave; jt < 16; jt += 8) {
    v8f c = {};
    #pragma unroll
    for (int kc = 0; kc < 2; ++kc)
      c = WMMA16(frag_a(Qs, DHH, kc * 32),
                 frag_b_t(k16 + h * DHH, INNER, kc * 32, jt * 16), c);
    store_c_lds(Ss + jt * 16, NN, c);
  }
  __syncthreads();

  // sparse bond corrections to logits: S[i][j] += q_i . (diff_ij @ We_h)
  for (int idx = t; idx < 2 * NB; idx += 256) {
    int bnd = idx >> 1, dir = idx & 1;
    int bi = bonds[bnd * 2 + 0], bj = bonds[bnd * 2 + 1];
    int i = dir ? bj : bi, j = dir ? bi : bj;
    if ((i >> 4) != it) continue;
    float sg = dir ? -1.f : 1.f;
    float dx = sg * (coords[bi * 3 + 0] - coords[bj * 3 + 0]);
    float dy = sg * (coords[bi * 3 + 1] - coords[bj * 3 + 1]);
    float dz = sg * (coords[bi * 3 + 2] - coords[bj * 3 + 2]);
    const _Float16* qi = Qs + (i & 15) * DHH;
    float corr = 0.f;
    #pragma unroll
    for (int d = 0; d < DHH; ++d) {
      float ev = dx * WeH[d] + dy * WeH[DHH + d] + dz * WeH[2 * DHH + d];
      corr += (float)qi[d] * ev;
    }
    atomicAdd(&Ss[(i & 15) * NN + j], corr);
  }
  __syncthreads();

  // softmax rows (2 rows per wave)
  #pragma unroll
  for (int rr = 0; rr < 2; ++rr) {
    int r = wave * 2 + rr;
    float mx = -3.4e38f;
    for (int j = lane; j < NN; j += 32) mx = fmaxf(mx, Ss[r * NN + j]);
    mx = wred_max(mx);
    float sm = 0.f;
    for (int j = lane; j < NN; j += 32) { float e = expf(Ss[r * NN + j] - mx); Ss[r * NN + j] = e; sm += e; }
    float inv = 1.f / wred_sum(sm);
    for (int j = lane; j < NN; j += 32) {
      float p = Ss[r * NN + j] * inv;
      Ss[r * NN + j] = p;
      P16[r * NN + j] = (_Float16)p;
    }
  }
  __syncthreads();

  // O = P @ V'  (16 x 64): v16T is d-major [d][j] -> B-friendly
  if (wave < 4) {
    int nt = wave;
    v8f c = {};
    #pragma unroll
    for (int kc = 0; kc < 8; ++kc)
      c = WMMA16(frag_a(P16, NN, kc * 32),
                 frag_b_t(v16T + (h * DHH) * NN, NN, kc * 32, nt * 16), c);
    store_c_lds(Os + nt * 16, DHH, c);
  }
  __syncthreads();

  // sparse bond corrections to output: O[i] += p_ij * (diff_ij @ We_h)
  for (int idx = t; idx < 2 * NB; idx += 256) {
    int bnd = idx >> 1, dir = idx & 1;
    int bi = bonds[bnd * 2 + 0], bj = bonds[bnd * 2 + 1];
    int i = dir ? bj : bi, j = dir ? bi : bj;
    if ((i >> 4) != it) continue;
    float sg = dir ? -1.f : 1.f;
    float dx = sg * (coords[bi * 3 + 0] - coords[bj * 3 + 0]);
    float dy = sg * (coords[bi * 3 + 1] - coords[bj * 3 + 1]);
    float dz = sg * (coords[bi * 3 + 2] - coords[bj * 3 + 2]);
    float p = Ss[(i & 15) * NN + j];
    #pragma unroll
    for (int d = 0; d < DHH; ++d) {
      float ev = dx * WeH[d] + dy * WeH[DHH + d] + dz * WeH[2 * DHH + d];
      atomicAdd(&Os[(i & 15) * DHH + d], p * ev);
    }
  }
  __syncthreads();

  for (int idx = t; idx < 16 * DHH / 4; idx += 256) {          // f32 LDS -> f16 global
    int r = idx >> 4, c4 = (idx & 15) * 4;
    float4 v = *(const float4*)&Os[r * DHH + c4];
    _Float16 tmp[4] = {(_Float16)v.x, (_Float16)v.y, (_Float16)v.z, (_Float16)v.w};
    *(uint64_t*)&attn16[(i0 + r) * INNER + h * DHH + c4] = *(const uint64_t*)tmp;
  }
}

// (A16[256x512] @ W[512x128] + bias), scalar-gated residual into nodes.
__global__ void k_proj_gate(const _Float16* __restrict__ Ah16,
                            const _Float16* __restrict__ WT, const float* __restrict__ bias,
                            const float* __restrict__ Wg, float* __restrict__ nodes) {
  __shared__ __align__(16) _Float16 A16[16 * INNER];
  __shared__ __align__(16) float    Ot[16 * DIMM];
  int i0 = blockIdx.x * 16, t = threadIdx.x, wave = t >> 5, lane = t & 31;
  for (int idx = t; idx < 16 * INNER / 8; idx += 256) {        // async A panel -> LDS
    int r = idx >> 6, c8 = (idx & 63) * 8;
    async_copy_b128(&A16[r * INNER + c8], &Ah16[(i0 + r) * INNER + c8]);
  }
  async_wait0();
  __syncthreads();
  {
    int nt = wave;                 // 8 waves, 8 column tiles of 16 -> 128 cols
    __builtin_prefetch(&WT[nt * 16 * INNER], 0, 1);
    v8f c = {};
    #pragma unroll
    for (int kc = 0; kc < 16; ++kc)
      c = WMMA16(frag_a(A16, INNER, kc * 32), frag_b_t(WT, INNER, kc * 32, nt * 16), c);
    int nn = lane & 15, hi = lane >> 4;
    union { v8f v; float f[8]; } u; u.v = c;
    #pragma unroll
    for (int r8 = 0; r8 < 8; ++r8)
      Ot[(r8 + hi * 8) * DIMM + nt * 16 + nn] = u.f[r8] + bias[nt * 16 + nn];
  }
  __syncthreads();
  #pragma unroll
  for (int rr = 0; rr < 2; ++rr) {
    int r = wave * 2 + rr, row = i0 + r;
    float acc = 0.f;
    for (int c = lane; c < DIMM; c += 32) {
      float o = Ot[r * DIMM + c], nd = nodes[row * DIMM + c];
      acc += o * Wg[c] + nd * Wg[DIMM + c] + (o - nd) * Wg[2 * DIMM + c];
    }
    acc = wred_sum(acc);
    float gt = 1.f / (1.f + expf(-acc));
    for (int c = lane; c < DIMM; c += 32) {
      float o = Ot[r * DIMM + c], nd = nodes[row * DIMM + c];
      nodes[row * DIMM + c] = o * gt + nd * (1.f - gt);
    }
  }
}

// LayerNorm2 + W1 + exact GeLU -> h16 (256 x 512, f16)
__global__ void k_ln_ffn1(const float* __restrict__ nodes,
                          const float* __restrict__ lg, const float* __restrict__ lb,
                          const _Float16* __restrict__ W1T, const float* __restrict__ b1,
                          _Float16* __restrict__ h16) {
  __shared__ __align__(16) float    Xs[16 * DIMM];
  __shared__ __align__(16) _Float16 Xh[16 * DIMM];
  int i0 = blockIdx.x * 16, t = threadIdx.x, wave = t >> 5, lane = t & 31;
  for (int idx = t; idx < 16 * DIMM / 4; idx += 256) {         // async f32 tile -> LDS
    int r = idx >> 5, c4 = (idx & 31) * 4;
    async_copy_b128(&Xs[r * DIMM + c4], &nodes[(i0 + r) * DIMM + c4]);
  }
  async_wait0();
  __syncthreads();
  #pragma unroll
  for (int rr = 0; rr < 2; ++rr) {
    int r = wave * 2 + rr;
    float s = 0.f;
    for (int c = lane; c < DIMM; c += 32) s += Xs[r * DIMM + c];
    float mean = wred_sum(s) * (1.f / DIMM);
    float vs = 0.f;
    for (int c = lane; c < DIMM; c += 32) { float d = Xs[r * DIMM + c] - mean; vs += d * d; }
    float inv = rsqrtf(wred_sum(vs) * (1.f / DIMM) + EPS);
    for (int c = lane; c < DIMM; c += 32)
      Xh[r * DIMM + c] = (_Float16)((Xs[r * DIMM + c] - mean) * inv * lg[c] + lb[c]);
  }
  __syncthreads();
  for (int tt = wave; tt < 32; tt += 8) {   // 512 cols = 32 tiles
    int n0 = tt * 16;
    __builtin_prefetch(&W1T[n0 * DIMM], 0, 1);
    v8f c = {};
    #pragma unroll
    for (int kc = 0; kc < 4; ++kc)
      c = WMMA16(frag_a(Xh, DIMM, kc * 32), frag_b_t(W1T, DIMM, kc * 32, n0), c);
    int nn = lane & 15, hi = lane >> 4;
    union { v8f v; float f[8]; } u; u.v = c;
    #pragma unroll
    for (int r8 = 0; r8 < 8; ++r8) {
      int row = i0 + r8 + hi * 8, col = n0 + nn;
      float y = u.f[r8] + b1[col];
      y = 0.5f * y * (1.f + erff(y * 0.70710678118f));  // exact gelu
      h16[row * 4 * DIMM + col] = (_Float16)y;
    }
  }
}

__global__ void k_energy(const float* __restrict__ nodes, const float* __restrict__ ow,
                         const float* __restrict__ ob, float* __restrict__ out) {
  int i = blockIdx.x * blockDim.x + threadIdx.x;
  if (i >= NN) return;
  float s = 0.f;
  #pragma unroll 4
  for (int c = 0; c < DIMM; ++c) s += nodes[i * DIMM + c] * ow[c];
  out[i] = s + ob[0];
}

// ---- host-side launch -----------------------------------------------------
extern "C" void kernel_launch(void* const* d_in, const int* in_sizes, int n_in,
                              void* d_out, int out_size, void* d_ws, size_t ws_size,
                              hipStream_t stream) {
  (void)in_sizes; (void)n_in; (void)out_size; (void)ws_size;
  const float* coords   = (const float*)d_in[0];
  const int*   bonds    = (const int*)  d_in[1];
  const float* noise    = (const float*)d_in[2];
  const float* atom_emb = (const float*)d_in[3];
  const float* ln1_g    = (const float*)d_in[4];
  const float* ln1_b    = (const float*)d_in[5];
  const float* Wq       = (const float*)d_in[6];
  const float* bq       = (const float*)d_in[7];
  const float* Wkv      = (const float*)d_in[8];
  const float* bkv      = (const float*)d_in[9];
  const float* We       = (const float*)d_in[10];
  const float* be       = (const float*)d_in[11];
  const float* Wo       = (const float*)d_in[12];
  const float* bo       = (const float*)d_in[13];
  const float* Wg1      = (const float*)d_in[14];
  const float* ln2_g    = (const float*)d_in[15];
  const float* ln2_b    = (const float*)d_in[16];
  const float* W1       = (const float*)d_in[17];
  const float* b1       = (const float*)d_in[18];
  const float* W2       = (const float*)d_in[19];
  const float* b2       = (const float*)d_in[20];
  const float* Wg2      = (const float*)d_in[21];
  const float* out_w    = (const float*)d_in[22];
  const float* out_b    = (const float*)d_in[23];

  char* ws = (char*)d_ws;
  float*    nodes  = (float*)ws;    ws += (size_t)NN * DIMM * 4;          // 128 KB
  _Float16* q16    = (_Float16*)ws; ws += (size_t)NN * INNER * 2;         // 256 KB
  _Float16* k16    = (_Float16*)ws; ws += (size_t)NN * INNER * 2;         // 256 KB
  _Float16* v16T   = (_Float16*)ws; ws += (size_t)NN * INNER * 2;         // 256 KB (d-major)
  _Float16* attn16 = (_Float16*)ws; ws += (size_t)NN * INNER * 2;         // 256 KB
  _Float16* h16    = (_Float16*)ws; ws += (size_t)NN * INNER * 2;         // 256 KB
  _Float16* WqT    = (_Float16*)ws; ws += (size_t)INNER * DIMM * 2;       // 128 KB
  _Float16* WkvT   = (_Float16*)ws; ws += (size_t)2 * INNER * DIMM * 2;   // 256 KB
  _Float16* WoT    = (_Float16*)ws; ws += (size_t)DIMM * INNER * 2;       // 128 KB
  _Float16* W1T    = (_Float16*)ws; ws += (size_t)4 * DIMM * DIMM * 2;    // 128 KB
  _Float16* W2T    = (_Float16*)ws; ws += (size_t)DIMM * 4 * DIMM * 2;    // 128 KB

  k_build_nodes<<<(NN * DIMM + 255) / 256, 256, 0, stream>>>(atom_emb, noise, nodes);

  for (int l = 0; l < DEPTH; ++l) {
    k_prep<<<1536, 256, 0, stream>>>(
        Wq + (size_t)l * DIMM * INNER, Wkv + (size_t)l * DIMM * 2 * INNER,
        Wo + (size_t)l * INNER * DIMM, W1 + (size_t)l * DIMM * 4 * DIMM,
        W2 + (size_t)l * 4 * DIMM * DIMM,
        WqT, WkvT, WoT, W1T, W2T);
    k_ln_qkv<<<16, 256, 0, stream>>>(nodes,
        ln1_g + l * DIMM, ln1_b + l * DIMM,
        WqT, bq + (size_t)l * INNER,
        WkvT, bkv + (size_t)l * 2 * INNER,
        be + (size_t)l * INNER,
        q16, k16, v16T);
    k_attn<<<HH * 16, 256, 0, stream>>>(q16, k16, v16T,
        We + (size_t)l * EDIM * INNER, bonds, coords, attn16);
    k_proj_gate<<<16, 256, 0, stream>>>(attn16,
        WoT, bo + (size_t)l * DIMM,
        Wg1 + (size_t)l * 3 * DIMM, nodes);
    k_ln_ffn1<<<16, 256, 0, stream>>>(nodes,
        ln2_g + l * DIMM, ln2_b + l * DIMM,
        W1T, b1 + (size_t)l * 4 * DIMM, h16);
    k_proj_gate<<<16, 256, 0, stream>>>(h16,
        W2T, b2 + (size_t)l * DIMM,
        Wg2 + (size_t)l * 3 * DIMM, nodes);
  }

  k_energy<<<1, 256, 0, stream>>>(nodes, out_w, out_b, (float*)d_out);
}
